// NaiveFourierKANLayer_3968549781589
// MI455X (gfx1250) — matrix operand order repaired and checked by reading, not compiled, not run on hardware
//
#include <hip/hip_runtime.h>
#include <hip/hip_bf16.h>
#include <stdint.h>

// FourierKAN forward as bf16 WMMA GEMM:
//   Y[B,J] = A[B,K] * W[K,J],  K = 2*I*G = 76800
//   A[b, 2*(i*G+g)+0] = cos(x[b,i]*(g+1)),  A[b, 2*(i*G+g)+1] = sin(...)
// Weights converted once fp32->bf16 into d_ws (78.6MB, L2-resident),
// layout Wb[j][kp] with kp = 2*(i*G+g)+d so one sincos feeds two K slots
// and each WMMA B-fragment lane is 16 contiguous bf16.

#define BSZ   4096
#define IDIM  128
#define JDIM  512
#define GS    300
#define KTOT  (2 * IDIM * GS)   // 76800
#define BK    32
#define NT    (KTOT / BK)       // 2400 K-tiles

#define BM 64
#define BN 128
#define NTHREADS 512            // 16 waves (wave32)

typedef __attribute__((ext_vector_type(16))) __bf16 v16bf;
typedef __attribute__((ext_vector_type(8)))  __bf16 v8bf;
typedef __attribute__((ext_vector_type(4)))  __bf16 v4bf;
typedef __attribute__((ext_vector_type(8)))  float  v8f;
typedef __attribute__((ext_vector_type(4)))  float  f32x4;
typedef __attribute__((ext_vector_type(4)))  unsigned int uint32x4;
typedef __attribute__((ext_vector_type(8)))  int int32x8;
typedef __attribute__((ext_vector_type(4)))  int int32x4;

// Tensor Data Mover path. This toolchain exposes the 6-arg builtin:
//   (uint32x4 g0, int32x8 g1, int32x4 g2, int32x4 g3, int32x8 extra, i32 cpol)
// Guarded so an absent builtin falls back to the proven synchronous staging.
#if defined(__HIP_DEVICE_COMPILE__) && \
    __has_builtin(__builtin_amdgcn_tensor_load_to_lds) && \
    __has_builtin(__builtin_amdgcn_s_wait_tensorcnt)
#define USE_TDM 1
#else
#define USE_TDM 0
#endif

__device__ __forceinline__ __bf16 f2bf(float f) {
    // round-to-nearest-even fp32 -> bf16
    unsigned u = __builtin_bit_cast(unsigned, f);
    unsigned r = (u + 0x7FFFu + ((u >> 16) & 1u)) >> 16;
    return __builtin_bit_cast(__bf16, (unsigned short)r);
}

// ---------------------------------------------------------------------------
// Kernel 0: convert fouriercoeffs[2][J][I][G] (fp32) -> Wb[j][kp] (bf16).
// ---------------------------------------------------------------------------
__global__ __launch_bounds__(256)
void wconvert(const float* __restrict__ fc, __bf16* __restrict__ wb) {
    size_t n = (size_t)blockIdx.x * 256 + threadIdx.x;   // < JDIM*KTOT
    int j   = (int)(n / KTOT);
    int kp  = (int)(n - (size_t)j * KTOT);
    int d   = kp & 1;
    int rem = kp >> 1;           // i*G + g
    int i   = rem / GS;
    int g   = rem - i * GS;
    float v = fc[(((size_t)d * JDIM + j) * IDIM + i) * GS + g];
    wb[n] = f2bf(v);
}

// ---------------------------------------------------------------------------
// Kernel 1: double-buffered WMMA GEMM; weight tiles staged by the Tensor
// Data Mover (one descriptor per K-step, issued by wave 0, TENSORcnt-tracked)
// overlapped with on-the-fly sincos activation generation.
// grid = (BSZ/BM, JDIM/BN) = (64, 4); 512 threads; 56KB LDS.
// ---------------------------------------------------------------------------
__global__ __launch_bounds__(NTHREADS)
void fkan_gemm(const float* __restrict__ x, const __bf16* __restrict__ wb,
               float* __restrict__ y) {
    __shared__ float  xs[BM][IDIM];      // 32KB staged x rows
    __shared__ __bf16 As[2][BM][BK];     // 8KB activation tiles (bf16)
    __shared__ __bf16 Bs[2][BN][BK];     // 16KB weight tiles (bf16)

    const int tid  = threadIdx.x;
    const int lane = tid & 31;
    const int wave = tid >> 5;        // 0..15
    const int wm   = wave & 3;        // M sub-tile (4 x 16 = BM)
    const int wn   = wave >> 2;       // N sub-tile (4 x 32 = BN)
    const int half = lane >> 4;
    const int l16  = lane & 15;

    const int mblk = blockIdx.x * BM;
    const int nblk = blockIdx.y * BN;

    // Stage x rows for this M-tile (coalesced 16B loads).
    {
        const f32x4* src = (const f32x4*)(x + (size_t)mblk * IDIM);
        f32x4* dst = (f32x4*)(&xs[0][0]);
#pragma unroll
        for (int t = 0; t < (BM * IDIM / 4) / NTHREADS; ++t)
            dst[tid + t * NTHREADS] = src[tid + t * NTHREADS];
    }
    __syncthreads();   // xs visible to activation generation

    // Activation generation: pair index ap is wave-uniform; rem = k0/2 + ap
    // advances by 16 per K-tile -> incremental (i,g), no div/mod in the loop.
    const int am = tid & (BM - 1);
    const int ap = (tid >> 6) * 2;           // 0,2,...,14
    int ii = 0, gg = ap;

    auto genA = [&](int nb) {
        float xf0 = xs[am][ii];
        int g1 = gg + 1, i1 = ii;
        if (g1 == GS) { g1 = 0; i1 = ii + 1; }
        float xf1 = (i1 == ii) ? xf0 : xs[am][i1];
        float s0, c0, s1, c1;
        __sincosf(xf0 * (float)(gg + 1), &s0, &c0);
        __sincosf(xf1 * (float)(g1 + 1), &s1, &c1);
        v4bf pk;
        pk[0] = f2bf(c0); pk[1] = f2bf(s0);
        pk[2] = f2bf(c1); pk[3] = f2bf(s1);
        *(v4bf*)(&As[nb][am][2 * ap]) = pk;   // one 8B ds_store
        gg += BK / 2;
        if (gg >= GS) { gg -= GS; ii++; }
    };

#if USE_TDM
    const unsigned ldsB[2] = { (unsigned)(uintptr_t)(&Bs[0][0][0]),
                               (unsigned)(uintptr_t)(&Bs[1][0][0]) };
    const uint64_t gbase = (uint64_t)(uintptr_t)(wb + (size_t)nblk * KTOT);
    // 2D tile: BN rows x BK elems (2B), row stride KTOT elems; D# per ISA 8.3/8.4.
    auto tdm_loadB = [&](int nb, int k0) {
        uint64_t ga = gbase + (uint64_t)k0 * 2;       // tile start (bytes)
        uint32x4 g0;
        g0.x = 1u;                                    // count=1, user D#
        g0.y = ldsB[nb];                              // lds_addr
        g0.z = (unsigned)ga;                          // global_addr[31:0]
        g0.w = (unsigned)(ga >> 32) | (2u << 30);     // addr[56:32] | type=2
        int32x8 g1;
        g1[0] = (1 << 16);                            // data_size=1 (2 bytes)
        g1[1] = (int)((KTOT & 0xFFFF) << 16);         // tensor_dim0[15:0]
        g1[2] = (KTOT >> 16) | (JDIM << 16);          // dim0[31:16]|dim1[15:0]
        g1[3] = (BK << 16);                           // dim1[31:16]=0|tile_dim0
        g1[4] = BN;                                   // tile_dim1, tile_dim2=0
        g1[5] = KTOT;                                 // tensor_dim0_stride lo
        g1[6] = 0;                                    // stride hi / dim1_stride
        g1[7] = 0;
        int32x4 gz4 = {0, 0, 0, 0};                   // groups 2/3 unused (2D)
        int32x8 gz8 = {0, 0, 0, 0, 0, 0, 0, 0};      // trailing group unused
        __builtin_amdgcn_tensor_load_to_lds(g0, g1, gz4, gz4, gz8, 0);
    };
#else
    const int bj = tid & (BN - 1);
    const int bk = (tid >> 7) * 8;                    // 0,8,16,24
    const __bf16* brow = wb + (size_t)(nblk + bj) * KTOT + bk;
#endif

    // Prologue: stage tile 0 of both operands.
#if USE_TDM
    if (wave == 0) tdm_loadB(0, 0);
#else
    *(v8bf*)(&Bs[0][bj][bk]) = *(const v8bf*)(brow);
#endif
    genA(0);
#if USE_TDM
    if (wave == 0) __builtin_amdgcn_s_wait_tensorcnt(0);
#endif
    __syncthreads();

    v8f acc0 = {}, acc1 = {};

    for (int t = 0; t < NT; ++t) {
        const int buf = t & 1, nb = buf ^ 1;
        const bool more = (t + 1) < NT;

        // Kick next weight tile DMA first: overlaps WMMA + sincos below.
        if (more) {
#if USE_TDM
            if (wave == 0) tdm_loadB(nb, (t + 1) * BK);
#else
            *(v8bf*)(&Bs[nb][bj][bk]) = *(const v8bf*)(brow + (t + 1) * BK);
            __builtin_prefetch(brow + (t + 2) * BK, 0, 0);  // global_prefetch_b8
#endif
        }

        // Fragments: contiguous k = 16*half + s in both A and B (same order, so valid).
        const __bf16* arow = &As[buf][wm * 16 + l16][0];
        v8bf alo = *(const v8bf*)(arow + half * 16);
        v8bf ahi = *(const v8bf*)(arow + half * 16 + 8);
        v16bf av = __builtin_shufflevector(alo, ahi,
                    0,1,2,3,4,5,6,7,8,9,10,11,12,13,14,15);

        const __bf16* bp0 = &Bs[buf][wn * 32 + l16][half * 16];
        v8bf b0lo = *(const v8bf*)(bp0);
        v8bf b0hi = *(const v8bf*)(bp0 + 8);
        v16bf bv0 = __builtin_shufflevector(b0lo, b0hi,
                    0,1,2,3,4,5,6,7,8,9,10,11,12,13,14,15);

        const __bf16* bp1 = &Bs[buf][wn * 32 + 16 + l16][half * 16];
        v8bf b1lo = *(const v8bf*)(bp1);
        v8bf b1hi = *(const v8bf*)(bp1 + 8);
        v16bf bv1 = __builtin_shufflevector(b1lo, b1hi,
                    0,1,2,3,4,5,6,7,8,9,10,11,12,13,14,15);

        acc0 = __builtin_amdgcn_wmma_f32_16x16x32_bf16(
                   false, av, false, bv0, (short)0, acc0, false, false);
        acc1 = __builtin_amdgcn_wmma_f32_16x16x32_bf16(
                   false, av, false, bv1, (short)0, acc1, false, false);

        if (more) {
            genA(nb);                     // trans-VALU overlaps the DMA
#if USE_TDM
            if (wave == 0) __builtin_amdgcn_s_wait_tensorcnt(0);
#endif
            __syncthreads();              // single barrier per K-tile
        }
    }

    // Epilogue: C/D layout — vgpr r holds M = r + 8*half, N = l16.
    const int orow = mblk + wm * 16 + half * 8;
    const int ocol = nblk + wn * 32 + l16;
#pragma unroll
    for (int r = 0; r < 8; ++r) {
        y[(size_t)(orow + r) * JDIM + ocol]      = acc0[r];
        y[(size_t)(orow + r) * JDIM + ocol + 16] = acc1[r];
    }
}

// ---------------------------------------------------------------------------
extern "C" void kernel_launch(void* const* d_in, const int* in_sizes, int n_in,
                              void* d_out, int out_size, void* d_ws, size_t ws_size,
                              hipStream_t stream) {
    const float* x  = (const float*)d_in[0];   // (B, I) fp32
    const float* fc = (const float*)d_in[1];   // (2, J, I, G) fp32
    float* y = (float*)d_out;                  // (B, J) fp32
    __bf16* wb = (__bf16*)d_ws;                // J*KTOT bf16 = 78.6 MB scratch

    // 1) weight conversion/reorder: JDIM*KTOT = 39,321,600 elems
    {
        dim3 grid((JDIM * (unsigned)KTOT) / 256);
        wconvert<<<grid, 256, 0, stream>>>(fc, wb);
    }
    // 2) fused activation + WMMA GEMM
    {
        dim3 grid(BSZ / BM, JDIM / BN);        // (64, 4)
        fkan_gemm<<<grid, NTHREADS, 0, stream>>>(x, wb, y);
    }
}
